// Attention_84533546320111
// MI455X (gfx1250) — compile-verified, matrix operand order checked by reading, not run on hardware
//
#include <hip/hip_runtime.h>
#include <hip/hip_bf16.h>

// ---------------------------------------------------------------------------
// MHA forward for MI455X (gfx1250), wave32, WMMA bf16 16x16x32, f32 accum.
// Register-blocked: GEMMs 32x64/wave (1.5 loads/WMMA, k-loop unrolled x2 for
// cross-iteration load/WMMA overlap), attention 32 query rows/wave
// (1 load/WMMA).
// ---------------------------------------------------------------------------

#define SEQ      2048
#define EMB      1024
#define NHEAD    16
#define HD       64
#define BS       4
#define NROWS    (BS * SEQ)          // 8192 flattened (b, s) rows

typedef __attribute__((ext_vector_type(16))) __bf16 v16bf;
typedef __attribute__((ext_vector_type(8)))  __bf16 v8bf;
typedef __attribute__((ext_vector_type(8)))  float  v8f;

// ---------------------------------------------------------------------------
// Fragment loader for the CDNA5 16-bit WMMA A/B layout (ISA 7.12.2):
//   lane l: row (row0 + l%16); element j<8 -> K = k0 + 8*(l/16) + j
//                              element j>=8 -> K = k0 + 16 + 8*(l/16) + (j-8)
// Two 16-byte loads per lane (global_load_b128 / ds_load_b128).
// ---------------------------------------------------------------------------
__device__ __forceinline__ v16bf load_frag16(const __bf16* base, int ld,
                                             int row0, int k0) {
  const int lane = threadIdx.x & 31;
  const int hf   = lane >> 4;        // 0 or 1
  const int r    = lane & 15;
  const __bf16* p = base + (size_t)(row0 + r) * (size_t)ld + k0 + hf * 8;
  v8bf lo = *(const v8bf*)(p);
  v8bf hi = *(const v8bf*)(p + 16);
  v16bf out;
#pragma unroll
  for (int i = 0; i < 8; ++i) { out[i] = lo[i]; out[i + 8] = hi[i]; }
  return out;
}

__device__ __forceinline__ v8f wmma_bf16(v16bf a, v16bf b, v8f c) {
  // D = A(16x32 bf16) x B(32x16 bf16) + C(16x16 f32)
  return __builtin_amdgcn_wmma_f32_16x16x32_bf16(
      /*neg_a=*/false, a, /*neg_b=*/false, b,
      /*c_mod=*/(short)0, c, /*reuse_a=*/false, /*reuse_b=*/false);
}

// ---------------------------------------------------------------------------
// fp32 -> bf16 conversion (grid-stride)
// ---------------------------------------------------------------------------
__global__ void cvt_bf16_kernel(const float* __restrict__ src,
                                __bf16* __restrict__ dst, int n) {
  int i      = blockIdx.x * blockDim.x + threadIdx.x;
  int stride = gridDim.x * blockDim.x;
  for (; i < n; i += stride) dst[i] = (__bf16)src[i];
}

// ---------------------------------------------------------------------------
// Fused QKV projection: y = x @ W^T + b  (M=8192, N=1024, K=1024)
// blockIdx.z selects {Q, K, V}. Q,K written [b,h,s,64]; V written transposed
// [b,h,64,s] so attention P*V B-fragments are contiguous.
// Wave computes a 32x64 register tile: per k-step 2 A-frags + 4 B-frags
// (12 b128 loads) feed 8 WMMAs; k-loop unrolled x2 to let the scheduler
// overlap next-step loads with current-step WMMAs.
// ---------------------------------------------------------------------------
__global__ __launch_bounds__(256) void qkv_gemm_kernel(
    const __bf16* __restrict__ xb,
    const __bf16* __restrict__ wqb, const __bf16* __restrict__ wkb,
    const __bf16* __restrict__ wvb,
    const float* __restrict__ bq, const float* __restrict__ bk,
    const float* __restrict__ bv,
    __bf16* __restrict__ qo, __bf16* __restrict__ ko,
    __bf16* __restrict__ vto) {
  const int wave = threadIdx.x >> 5;
  const int lane = threadIdx.x & 31;
  const int hf   = lane >> 4;
  const int nl   = lane & 15;
  const int wy   = wave >> 1;                 // 4 M positions
  const int wx   = wave & 1;                  // 2 N positions

  const int m0    = (blockIdx.y * 4 + wy) * 32;
  const int n0    = (blockIdx.x * 2 + wx) * 64;
  const int which = blockIdx.z;

  const __bf16* w    = (which == 0) ? wqb : (which == 1) ? wkb : wvb;
  const float*  bias = (which == 0) ? bq  : (which == 1) ? bk  : bv;

  v8f acc[2][4];
#pragma unroll
  for (int i = 0; i < 2; ++i)
#pragma unroll
    for (int j = 0; j < 4; ++j)
#pragma unroll
      for (int r = 0; r < 8; ++r) acc[i][j][r] = 0.0f;

#pragma unroll 2
  for (int k0 = 0; k0 < EMB; k0 += 32) {
    v16bf a[2], b[4];
#pragma unroll
    for (int i = 0; i < 2; ++i) a[i] = load_frag16(xb, EMB, m0 + 16 * i, k0);
#pragma unroll
    for (int j = 0; j < 4; ++j) b[j] = load_frag16(w, EMB, n0 + 16 * j, k0);
#pragma unroll
    for (int i = 0; i < 2; ++i)
#pragma unroll
      for (int j = 0; j < 4; ++j)
        acc[i][j] = wmma_bf16(a[i], b[j], acc[i][j]);
  }

#pragma unroll
  for (int j = 0; j < 4; ++j) {
    const int   col  = n0 + 16 * j + nl;
    const float bval = bias[col];
    const int   head = col >> 6;
    const int   d    = col & 63;
#pragma unroll
    for (int i = 0; i < 2; ++i) {
#pragma unroll
      for (int r = 0; r < 8; ++r) {
        const int   row = m0 + 16 * i + r + 8 * hf;  // flattened (b, s)
        const int   bb  = row >> 11;
        const int   s   = row & (SEQ - 1);
        const float val = acc[i][j][r] + bval;
        if (which == 2) {
          vto[(((size_t)bb * NHEAD + head) * HD + d) * SEQ + s] = (__bf16)val;
        } else {
          __bf16* dst = (which == 0) ? qo : ko;
          dst[(((size_t)bb * NHEAD + head) * SEQ + s) * HD + d] = (__bf16)val;
        }
      }
    }
  }
}

// ---------------------------------------------------------------------------
// Flash attention: one wave owns a 32-row query tile (2 x 16-row sub-tiles
// sharing every K and V fragment); iterates 32 keys/step.
// Per step: 8 score WMMAs + 8 P*V WMMAs against 16 b128 loads (1 load/WMMA).
// Online softmax via shfl_xor row reductions (rows live in 16-lane halves
// per the C-matrix layout); P re-shaped C-layout -> A-fragment via per-wave
// LDS tiles.
// ---------------------------------------------------------------------------
__global__ __launch_bounds__(256) void attn_kernel(
    const __bf16* __restrict__ qb, const __bf16* __restrict__ kb,
    const __bf16* __restrict__ vtb, __bf16* __restrict__ ob) {
  // per-wave P staging: 2 sub-tiles x 16 rows x 32 cols bf16, row stride 40
  // (80B rows: 16B aligned, spreads LDS banks). 8 waves -> 20 KB.
  __shared__ __align__(16) __bf16 pbuf[8][2 * 16 * 40];

  const int wave = threadIdx.x >> 5;
  const int lane = threadIdx.x & 31;
  const int hf   = lane >> 4;
  const int nl   = lane & 15;

  const int q0   = (blockIdx.x * 8 + wave) * 32;
  const int head = blockIdx.y;
  const int bb   = blockIdx.z;

  const __bf16* qh = qb  + ((size_t)bb * NHEAD + head) * SEQ * HD;
  const __bf16* kh = kb  + ((size_t)bb * NHEAD + head) * SEQ * HD;
  const __bf16* vh = vtb + ((size_t)bb * NHEAD + head) * HD * SEQ;

  // Q fragments held in registers for the whole pass: [q-subtile][k-half]
  v16bf qa[2][2];
#pragma unroll
  for (int qt = 0; qt < 2; ++qt) {
    qa[qt][0] = load_frag16(qh, HD, q0 + 16 * qt, 0);
    qa[qt][1] = load_frag16(qh, HD, q0 + 16 * qt, 32);
  }

  float mrun[2][8], lrun[2][8];
  v8f acc[2][4];
#pragma unroll
  for (int qt = 0; qt < 2; ++qt) {
#pragma unroll
    for (int r = 0; r < 8; ++r) { mrun[qt][r] = -3.0e38f; lrun[qt][r] = 0.0f; }
#pragma unroll
    for (int t = 0; t < 4; ++t)
#pragma unroll
      for (int r = 0; r < 8; ++r) acc[qt][t][r] = 0.0f;
  }

  const float scale = 0.125f;  // 1/sqrt(64)
  __bf16* pl = &pbuf[wave][0];

  for (int kt = 0; kt < SEQ; kt += 32) {
    // ---- K fragments (shared by both query sub-tiles) ----
    v16bf kf[4];
    kf[0] = load_frag16(kh, HD, kt,      0);
    kf[1] = load_frag16(kh, HD, kt,      32);
    kf[2] = load_frag16(kh, HD, kt + 16, 0);
    kf[3] = load_frag16(kh, HD, kt + 16, 32);

#pragma unroll
    for (int qt = 0; qt < 2; ++qt) {
      v8f s0 = {}, s1 = {};
      s0 = wmma_bf16(qa[qt][0], kf[0], s0);
      s0 = wmma_bf16(qa[qt][1], kf[1], s0);
      s1 = wmma_bf16(qa[qt][0], kf[2], s1);
      s1 = wmma_bf16(qa[qt][1], kf[3], s1);
#pragma unroll
      for (int r = 0; r < 8; ++r) { s0[r] *= scale; s1[r] *= scale; }

      // ---- online softmax stats (reduce across 16-lane half) ----
      float mnew[8], corr[8], p0[8], p1[8];
#pragma unroll
      for (int r = 0; r < 8; ++r) {
        float t = fmaxf(s0[r], s1[r]);
#pragma unroll
        for (int off = 8; off >= 1; off >>= 1)
          t = fmaxf(t, __shfl_xor(t, off, 32));
        mnew[r] = fmaxf(mrun[qt][r], t);
        corr[r] = __expf(mrun[qt][r] - mnew[r]);
        mrun[qt][r] = mnew[r];
      }
#pragma unroll
      for (int r = 0; r < 8; ++r) {
        p0[r] = __expf(s0[r] - mnew[r]);
        p1[r] = __expf(s1[r] - mnew[r]);
        float rs = p0[r] + p1[r];
#pragma unroll
        for (int off = 8; off >= 1; off >>= 1)
          rs += __shfl_xor(rs, off, 32);
        lrun[qt][r] = lrun[qt][r] * corr[r] + rs;
      }
#pragma unroll
      for (int t = 0; t < 4; ++t)
#pragma unroll
        for (int r = 0; r < 8; ++r) acc[qt][t][r] *= corr[r];

      // ---- stage P through LDS: C-layout -> row-major 16x32 bf16 ----
      __bf16* pq = pl + qt * 16 * 40;
#pragma unroll
      for (int r = 0; r < 8; ++r) {
        const int m = r + 8 * hf;
        pq[m * 40 + nl]      = (__bf16)p0[r];
        pq[m * 40 + 16 + nl] = (__bf16)p1[r];
      }
    }

    // same-wave LDS ops are in-order (DScnt); compiler inserts waits.
    v16bf pa0 = load_frag16(pl,           40, 0, 0);
    v16bf pa1 = load_frag16(pl + 16 * 40, 40, 0, 0);

    // ---- O(32x64) += P(32x32) @ V(32x64); V frag feeds both sub-tiles ----
#pragma unroll
    for (int t = 0; t < 4; ++t) {
      v16bf vf = load_frag16(vh, SEQ, t * 16, kt);
      acc[0][t] = wmma_bf16(pa0, vf, acc[0][t]);
      acc[1][t] = wmma_bf16(pa1, vf, acc[1][t]);
    }
  }

  // ---- normalize and store attention output [b, s, h*64+d] as bf16 ----
#pragma unroll
  for (int qt = 0; qt < 2; ++qt) {
    float inv[8];
#pragma unroll
    for (int r = 0; r < 8; ++r) inv[r] = 1.0f / lrun[qt][r];
#pragma unroll
    for (int t = 0; t < 4; ++t) {
#pragma unroll
      for (int r = 0; r < 8; ++r) {
        const int    m   = r + 8 * hf;
        const size_t row = (size_t)bb * SEQ + q0 + 16 * qt + m;
        ob[row * EMB + head * HD + t * 16 + nl] =
            (__bf16)(acc[qt][t][r] * inv[r]);
      }
    }
  }
}

// ---------------------------------------------------------------------------
// Output projection: out = attn @ Wo^T + bo  (f32 store), 32x64 wave tile.
// ---------------------------------------------------------------------------
__global__ __launch_bounds__(256) void out_gemm_kernel(
    const __bf16* __restrict__ ab, const __bf16* __restrict__ wob,
    const float* __restrict__ bo, float* __restrict__ out) {
  const int wave = threadIdx.x >> 5;
  const int lane = threadIdx.x & 31;
  const int hf   = lane >> 4;
  const int nl   = lane & 15;
  const int wy   = wave >> 1;
  const int wx   = wave & 1;

  const int m0 = (blockIdx.y * 4 + wy) * 32;
  const int n0 = (blockIdx.x * 2 + wx) * 64;

  v8f acc[2][4];
#pragma unroll
  for (int i = 0; i < 2; ++i)
#pragma unroll
    for (int j = 0; j < 4; ++j)
#pragma unroll
      for (int r = 0; r < 8; ++r) acc[i][j][r] = 0.0f;

#pragma unroll 2
  for (int k0 = 0; k0 < EMB; k0 += 32) {
    v16bf a[2], b[4];
#pragma unroll
    for (int i = 0; i < 2; ++i) a[i] = load_frag16(ab, EMB, m0 + 16 * i, k0);
#pragma unroll
    for (int j = 0; j < 4; ++j) b[j] = load_frag16(wob, EMB, n0 + 16 * j, k0);
#pragma unroll
    for (int i = 0; i < 2; ++i)
#pragma unroll
      for (int j = 0; j < 4; ++j)
        acc[i][j] = wmma_bf16(a[i], b[j], acc[i][j]);
  }

#pragma unroll
  for (int j = 0; j < 4; ++j) {
    const int   col  = n0 + 16 * j + nl;
    const float bval = bo[col];
#pragma unroll
    for (int i = 0; i < 2; ++i) {
#pragma unroll
      for (int r = 0; r < 8; ++r) {
        const int row = m0 + 16 * i + r + 8 * hf;
        out[(size_t)row * EMB + col] = acc[i][j][r] + bval;
      }
    }
  }
}

// ---------------------------------------------------------------------------
// Host launcher
// ---------------------------------------------------------------------------
extern "C" void kernel_launch(void* const* d_in, const int* in_sizes, int n_in,
                              void* d_out, int out_size, void* d_ws,
                              size_t ws_size, hipStream_t stream) {
  const float* x  = (const float*)d_in[0];
  const float* wq = (const float*)d_in[1];
  const float* bq = (const float*)d_in[2];
  const float* wk = (const float*)d_in[3];
  const float* bk = (const float*)d_in[4];
  const float* wv = (const float*)d_in[5];
  const float* bv = (const float*)d_in[6];
  const float* wo = (const float*)d_in[7];
  const float* bo = (const float*)d_in[8];
  float* out = (float*)d_out;

  // workspace layout (bytes); total = 88 MiB
  char* ws = (char*)d_ws;
  const size_t SZ_X  = (size_t)NROWS * EMB * sizeof(__bf16);   // 16 MiB
  const size_t SZ_W  = (size_t)EMB * EMB * sizeof(__bf16);     //  2 MiB
  const size_t SZ_T  = (size_t)BS * NHEAD * SEQ * HD * sizeof(__bf16); // 16 MiB
  __bf16* xb  = (__bf16*)(ws);
  __bf16* wqb = (__bf16*)(ws + SZ_X);
  __bf16* wkb = (__bf16*)(ws + SZ_X + SZ_W);
  __bf16* wvb = (__bf16*)(ws + SZ_X + 2 * SZ_W);
  __bf16* wob = (__bf16*)(ws + SZ_X + 3 * SZ_W);
  __bf16* qb  = (__bf16*)(ws + SZ_X + 4 * SZ_W);
  __bf16* kb  = (__bf16*)(ws + SZ_X + 4 * SZ_W + SZ_T);
  __bf16* vtb = (__bf16*)(ws + SZ_X + 4 * SZ_W + 2 * SZ_T);
  __bf16* ob  = (__bf16*)(ws + SZ_X + 4 * SZ_W + 3 * SZ_T);

  // 1) fp32 -> bf16 conversions
  cvt_bf16_kernel<<<4096, 256, 0, stream>>>(x, xb, NROWS * EMB);
  cvt_bf16_kernel<<<1024, 256, 0, stream>>>(wq, wqb, EMB * EMB);
  cvt_bf16_kernel<<<1024, 256, 0, stream>>>(wk, wkb, EMB * EMB);
  cvt_bf16_kernel<<<1024, 256, 0, stream>>>(wv, wvb, EMB * EMB);
  cvt_bf16_kernel<<<1024, 256, 0, stream>>>(wo, wob, EMB * EMB);

  // 2) fused QKV projection (z: 0=Q, 1=K, 2=V-transposed)
  //    block = 8 waves covering 128 (M) x 128 (N)
  dim3 gqkv(EMB / 128, NROWS / 128, 3);
  qkv_gemm_kernel<<<gqkv, 256, 0, stream>>>(xb, wqb, wkb, wvb, bq, bk, bv,
                                            qb, kb, vtb);

  // 3) flash attention (one wave per 32-row query tile)
  dim3 gat(SEQ / 256, NHEAD, BS);
  attn_kernel<<<gat, 256, 0, stream>>>(qb, kb, vtb, ob);

  // 4) output projection (f32 result)
  dim3 gout(EMB / 128, NROWS / 128, 1);
  out_gemm_kernel<<<gout, 256, 0, stream>>>(ob, wob, bo, out);

  (void)in_sizes; (void)n_in; (void)out_size; (void)ws_size;
}
